// ConvMultiStepAttention_27865747816848
// MI455X (gfx1250) — compile-verified, matrix operand order checked by reading, not run on hardware
//
#include <hip/hip_runtime.h>
#include <hip/hip_bf16.h>

// ---------------------------------------------------------------------------
// Problem constants (from reference): B=8, C=512, T=2048, S=2048
// ---------------------------------------------------------------------------
#define BB 8
#define CC 512
#define TT 2048
#define SS 2048
#define SCALE_W 0.70710678118654752440f

typedef __attribute__((ext_vector_type(16))) __bf16 v16bf;
typedef __attribute__((ext_vector_type(8)))  __bf16 v8bf;
typedef __attribute__((ext_vector_type(8)))  float  v8f;

// ---------------------------------------------------------------------------
// Fragment loader: 16x32 bf16 tile from row-major [rows, K] storage, ld = K.
// CDNA5 16-bit A/B layout: lane L holds row (L&15); K-halves by (L>>4):
//   lanes 0-15 : K = 0..7  and 16..23
//   lanes 16-31: K = 8..15 and 24..31
// => two contiguous 16-byte loads per lane.
// ---------------------------------------------------------------------------
__device__ __forceinline__ v16bf load_frag(const __bf16* __restrict__ p, int ld) {
    int lane = threadIdx.x & 31;
    int r    = lane & 15;
    int kb   = (lane >> 4) << 3;      // 0 or 8
    const __bf16* q = p + (size_t)r * ld + kb;
    v8bf lo = *(const v8bf*)(q);      // K = kb .. kb+7
    v8bf hi = *(const v8bf*)(q + 16); // K = kb+16 .. kb+23
    return __builtin_shufflevector(lo, hi, 0,1,2,3,4,5,6,7,8,9,10,11,12,13,14,15);
}

// ---------------------------------------------------------------------------
// Elementwise fp32 -> bf16 convert
// ---------------------------------------------------------------------------
__global__ void k_cvt_bf16(const float* __restrict__ in, __bf16* __restrict__ out, int n) {
    int i = blockIdx.x * blockDim.x + threadIdx.x;
    if (i < n) out[i] = (__bf16)in[i];
}

// ---------------------------------------------------------------------------
// Tiled transpose + convert: in [B, rows, cols] f32 -> out [B, cols, rows] bf16
// block (32,8), grid (cols/32, rows/32, B)
// ---------------------------------------------------------------------------
__global__ void k_transpose_cvt(const float* __restrict__ in, __bf16* __restrict__ out,
                                int rows, int cols) {
    __shared__ float tile[32][33];
    int b  = blockIdx.z;
    int r0 = blockIdx.y * 32;
    int c0 = blockIdx.x * 32;
    const float* src = in  + (size_t)b * rows * cols;
    __bf16*      dst = out + (size_t)b * rows * cols;
    int tx = threadIdx.x, ty = threadIdx.y;
#pragma unroll
    for (int i = 0; i < 32; i += 8)
        tile[ty + i][tx] = src[(size_t)(r0 + ty + i) * cols + c0 + tx];
    __syncthreads();
#pragma unroll
    for (int i = 0; i < 32; i += 8)
        dst[(size_t)(c0 + ty + i) * rows + r0 + tx] = (__bf16)tile[tx][ty + i];
}

// ---------------------------------------------------------------------------
// GEMM1: targetT[b,t,o] = (base[b,o,t] + bias[o] + sum_c W[o,c]*x[b,c,t]) * SCALE
// A = xT [t,c] (M=t), B = Wbf [o,c] (N=o), K = C. One 16x16 tile per wave.
// 256 threads = 8 waves/block; tiles: B * (T/16) * (C/16) = 8*128*32 = 32768.
// ---------------------------------------------------------------------------
__global__ __launch_bounds__(256)
void k_gemm1(const __bf16* __restrict__ xT, const __bf16* __restrict__ Wbf,
             const float* __restrict__ base, const float* __restrict__ bias,
             __bf16* __restrict__ targetT) {
    int wave = threadIdx.x >> 5;
    int tile = blockIdx.x * 8 + wave;
    int ot = tile & 31;           // C/16 = 32
    int tt = (tile >> 5) & 127;   // T/16 = 128
    int b  = tile >> 12;
    int t0 = tt * 16, o0 = ot * 16;

    const __bf16* A  = xT  + ((size_t)b * TT + t0) * CC;
    const __bf16* Bp = Wbf + (size_t)o0 * CC;

    v8f acc = {};
#pragma unroll
    for (int k = 0; k < CC; k += 32) {
        v16bf a = load_frag(A + k, CC);
        v16bf w = load_frag(Bp + k, CC);
        acc = __builtin_amdgcn_wmma_f32_16x16x32_bf16(false, a, false, w,
                                                      (short)0, acc, false, false);
    }

    int lane = threadIdx.x & 31;
    int n    = lane & 15;            // o within tile
    int mh   = (lane >> 4) << 3;     // row-half
    int o    = o0 + n;
    float bo = bias[o];
#pragma unroll
    for (int r = 0; r < 8; ++r) {
        int t = t0 + r + mh;
        float v = (acc[r] + bo + base[((size_t)b * CC + o) * TT + t]) * SCALE_W;
        targetT[((size_t)b * TT + t) * CC + o] = (__bf16)v;
    }
}

// ---------------------------------------------------------------------------
// Attention: per block -> 16 rows of t, full S.
//   scores[t,s] = sum_c targetT[b,t,c] * eotT[b,s,c]   (A frags held in regs)
//   softmax over s, write fp32 attn (output) + bf16 attn (workspace)
// Dynamic LDS: 16*S floats for scores + 16*17 for reductions.
// ---------------------------------------------------------------------------
__global__ __launch_bounds__(256)
void k_attn(const __bf16* __restrict__ targetT, const __bf16* __restrict__ eotT,
            float* __restrict__ attn_out, __bf16* __restrict__ attn_bf) {
    extern __shared__ float smem[];
    float* scores = smem;               // [16][S]
    float* red    = smem + 16 * SS;     // [16][17]

    int blk = blockIdx.x;
    int tt  = blk & 127;                // T/16 = 128
    int b   = blk >> 7;
    int t0  = tt * 16;
    int wave = threadIdx.x >> 5;
    int lane = threadIdx.x & 31;

    const __bf16* A     = targetT + ((size_t)b * TT + t0) * CC;
    const __bf16* Bbase = eotT    + (size_t)b * SS * CC;

    // Preload all 16 K-fragments of A (K = C = 512) into registers.
    v16bf afrag[16];
#pragma unroll
    for (int k = 0; k < 16; ++k) afrag[k] = load_frag(A + k * 32, CC);

    int n  = lane & 15;
    int mh = (lane >> 4) << 3;

    for (int si = wave; si < SS / 16; si += 8) {
        int s0 = si * 16;
        v8f acc = {};
#pragma unroll
        for (int k = 0; k < 16; ++k) {
            v16bf bfrg = load_frag(Bbase + (size_t)s0 * CC + k * 32, CC);
            acc = __builtin_amdgcn_wmma_f32_16x16x32_bf16(false, afrag[k], false, bfrg,
                                                          (short)0, acc, false, false);
        }
#pragma unroll
        for (int r = 0; r < 8; ++r)
            scores[(r + mh) * SS + s0 + n] = acc[r];
    }
    __syncthreads();

    // Softmax: 16 threads per row; thread g of row handles cols {i*16 + g}.
    int row = threadIdx.x >> 4;   // 0..15
    int g   = threadIdx.x & 15;   // 0..15

    float mx = -1e30f;
    for (int i = 0; i < SS / 16; ++i)
        mx = fmaxf(mx, scores[row * SS + i * 16 + g]);
    red[row * 17 + g] = mx;
    __syncthreads();
    if (g == 0) {
        float m = red[row * 17 + 0];
#pragma unroll
        for (int i = 1; i < 16; ++i) m = fmaxf(m, red[row * 17 + i]);
        red[row * 17 + 16] = m;
    }
    __syncthreads();
    float rowmax = red[row * 17 + 16];

    float sum = 0.f;
    for (int i = 0; i < SS / 16; ++i) {
        float e = __expf(scores[row * SS + i * 16 + g] - rowmax);
        scores[row * SS + i * 16 + g] = e;
        sum += e;
    }
    __syncthreads();
    red[row * 17 + g] = sum;
    __syncthreads();
    if (g == 0) {
        float s = 0.f;
#pragma unroll
        for (int i = 0; i < 16; ++i) s += red[row * 17 + i];
        red[row * 17 + 16] = s;
    }
    __syncthreads();
    float inv = 1.0f / red[row * 17 + 16];

    float*  orow = attn_out + ((size_t)b * TT + t0 + row) * SS;
    __bf16* brow = attn_bf  + ((size_t)b * TT + t0 + row) * SS;
    for (int i = 0; i < SS / 16; ++i) {
        float v = scores[row * SS + i * 16 + g] * inv;
        orow[i * 16 + g] = v;
        brow[i * 16 + g] = (__bf16)v;
    }
}

// ---------------------------------------------------------------------------
// GEMM3: out[b,c,t] = sum_s attn[b,t,s] * eoc[b,c,s]
// A = eocbf [c,s] (M=c), B = attnbf [t,s] (N=t) -> D stores contiguous in t.
// Block = 8 waves; wave w covers c in [w*64, w*64+64) via 4 accumulators
// sharing each B fragment. Grid = B * T/16 = 1024 blocks.
// ---------------------------------------------------------------------------
__global__ __launch_bounds__(256)
void k_ctx(const __bf16* __restrict__ attn_bf, const __bf16* __restrict__ eocbf,
           float* __restrict__ out) {
    int blk = blockIdx.x;
    int tt  = blk & 127;
    int b   = blk >> 7;
    int t0  = tt * 16;
    int wave = threadIdx.x >> 5;
    int lane = threadIdx.x & 31;
    int c0 = wave * 64;

    const __bf16* Bp = attn_bf + ((size_t)b * TT + t0) * SS;
    const __bf16* Ap = eocbf   + (size_t)b * CC * SS;

    v8f acc[4] = {};
    for (int k = 0; k < SS; k += 32) {
        v16bf bfrg = load_frag(Bp + k, SS);
#pragma unroll
        for (int mi = 0; mi < 4; ++mi) {
            v16bf afrg = load_frag(Ap + (size_t)(c0 + mi * 16) * SS + k, SS);
            acc[mi] = __builtin_amdgcn_wmma_f32_16x16x32_bf16(false, afrg, false, bfrg,
                                                              (short)0, acc[mi], false, false);
        }
    }

    int n  = lane & 15;          // t within tile
    int mh = (lane >> 4) << 3;   // c row half
#pragma unroll
    for (int mi = 0; mi < 4; ++mi) {
#pragma unroll
        for (int r = 0; r < 8; ++r) {
            int c = c0 + mi * 16 + r + mh;
            out[((size_t)b * CC + c) * TT + t0 + n] = acc[mi][r];
        }
    }
}

// ---------------------------------------------------------------------------
// Host-side launch
// ---------------------------------------------------------------------------
extern "C" void kernel_launch(void* const* d_in, const int* in_sizes, int n_in,
                              void* d_out, int out_size, void* d_ws, size_t ws_size,
                              hipStream_t stream) {
    const float* base = (const float*)d_in[0];   // [B,C,T,1]
    const float* x    = (const float*)d_in[1];   // [B,C,T,1]
    const float* eot  = (const float*)d_in[2];   // [B,C,S]
    const float* eoc  = (const float*)d_in[3];   // [B,C,S]
    const float* W    = (const float*)d_in[4];   // [C,C]
    const float* bias = (const float*)d_in[5];   // [C]

    float* ctx_out  = (float*)d_out;                               // [B,C,T]
    float* attn_out = (float*)d_out + (size_t)BB * CC * TT;        // [B,T,S]

    // Workspace layout (bytes)
    char* ws = (char*)d_ws;
    __bf16* Wbf     = (__bf16*)(ws);                                   // 512 KB
    __bf16* xT      = (__bf16*)(ws + (512u << 10));                    // 16 MB
    __bf16* eotT    = (__bf16*)(ws + (512u << 10) + (16u << 20));      // 16 MB
    __bf16* eocbf   = (__bf16*)(ws + (512u << 10) + (32u << 20));      // 16 MB
    __bf16* targetT = (__bf16*)(ws + (512u << 10) + (48u << 20));      // 16 MB
    __bf16* attn_bf = (__bf16*)(ws + (512u << 10) + (64u << 20));      // 64 MB

    // --- converts / transposes ---
    {
        int n = CC * CC;
        k_cvt_bf16<<<(n + 255) / 256, 256, 0, stream>>>(W, Wbf, n);
    }
    {
        dim3 g(TT / 32, CC / 32, BB), blk(32, 8);
        k_transpose_cvt<<<g, blk, 0, stream>>>(x, xT, CC, TT);
    }
    {
        dim3 g(SS / 32, CC / 32, BB), blk(32, 8);
        k_transpose_cvt<<<g, blk, 0, stream>>>(eot, eotT, CC, SS);
    }
    {
        int n = BB * CC * SS;
        k_cvt_bf16<<<(n + 255) / 256, 256, 0, stream>>>(eoc, eocbf, n);
    }

    // --- GEMM1 + epilogue -> targetT ---
    k_gemm1<<<(BB * (TT / 16) * (CC / 16)) / 8, 256, 0, stream>>>(
        xT, Wbf, base, bias, targetT);

    // --- scores + softmax -> attn (fp32 out + bf16 ws) ---
    size_t lds_bytes = (size_t)(16 * SS + 16 * 17) * sizeof(float);
    k_attn<<<BB * (TT / 16), 256, lds_bytes, stream>>>(targetT, eotT, attn_out, attn_bf);

    // --- context GEMM -> ctx_out [B,C,T] ---
    k_ctx<<<BB * (TT / 16), 256, 0, stream>>>(attn_bf, eocbf, ctx_out);
}
